// MultiConvAttn_66967130079885
// MI455X (gfx1250) — matrix operand order repaired and checked by reading, not compiled, not run on hardware
//
#include <hip/hip_runtime.h>
#include <hip/hip_bf16.h>

// ---------------------------------------------------------------------------
// MultiConvAttn on MI455X (gfx1250, wave32, WMMA).
//
// Math: softmax over keys of (sq[j] + sk[k]) is independent of j (sq cancels),
// so out[b,j,:,h,w] = sum_k softmax_k(sk)[k] * K[b,k,:,h,w], identical for all
// j. Only the K-conv matters; it runs as implicit GEMM on
// v_wmma_f32_16x16x32_bf16 with a bf16 hi/lo split (3 WMMAs) for ~fp32
// accuracy with f32 accumulation.
//
// This revision: each wave owns ONE cout tile (mtile) and TWO pixel tiles,
// holding two accumulators so every A hi/lo fragment pair (global loads)
// feeds 6 WMMAs instead of 3 — halves A-side global traffic per WMMA.
// ---------------------------------------------------------------------------

typedef __attribute__((ext_vector_type(16))) __bf16        v16bf;
typedef __attribute__((ext_vector_type(8)))  float         v8f;
typedef __attribute__((ext_vector_type(8)))  unsigned int  v8u;
typedef __attribute__((ext_vector_type(4)))  unsigned int  v4u;
__device__ __forceinline__ v8u ldB(const __bf16* p) {
  v4u a = *(const v4u*)p, b = *(const v4u*)(p + 16);
  v8u r; r.lo = a; r.hi = b; return r;
}

#define NHEADS 4
#define CDIM   16      // head dim
#define DMODEL 64
#define NB     4
#define NS     32
#define NIMG   128     // NB*NS
#define HW     64      // H == W == 64
#define KSTEPS 18      // 9 taps * (64 cin / 32 per wmma)

// ---- bf16 helpers (round-to-nearest-even truncation + residual) -----------
__device__ __forceinline__ unsigned short f2bf_rne(float v) {
  unsigned u = __float_as_uint(v);
  unsigned r = u + 0x7FFFu + ((u >> 16) & 1u);
  return (unsigned short)(r >> 16);
}
__device__ __forceinline__ float bf2f(unsigned short b) {
  return __uint_as_float(((unsigned)b) << 16);
}

// ---------------------------------------------------------------------------
// Kernel 0: swizzle Wk (flat [64 cout][64 cin][3][3] f32) into A-fragment
// order, bf16 hi + lo.  A-layout (16-bit A 16x32, ISA 7.12.2):
//   lane<16 : M=lane,    K = {0..7, 16..23}
//   lane>=16: M=lane-16, K = {8..15, 24..31}
// Storage: AHI/ALO[(mtile*18 + t)*32 + lane][16 bf16]  (32 B/lane, coalesced)
// K-step t = tap*2 + half, tap=(ky,kx), half selects cin base {0,32}.
// ---------------------------------------------------------------------------
__global__ void wk_rearrange(const float* __restrict__ Wk,
                             __bf16* __restrict__ AHI,
                             __bf16* __restrict__ ALO) {
  int tid = blockIdx.x * blockDim.x + threadIdx.x;
  if (tid >= 4 * KSTEPS * 32) return;
  int lane  = tid & 31;
  int t     = (tid >> 5) % KSTEPS;
  int mtile = tid / (KSTEPS * 32);
  int tap = t >> 1, half = t & 1;
  int ky = tap / 3, kx = tap % 3;
  int cb = half * 32;
  int cout = mtile * 16 + (lane & 15);
  int kadd = (lane >= 16) ? 8 : 0;
  int obase = ((mtile * KSTEPS + t) * 32 + lane) * 16;
#pragma unroll
  for (int j = 0; j < 16; ++j) {
    int kk  = ((j < 8) ? j : j + 8) + kadd;      // {0..7,16..23} / {8..15,24..31}
    int cin = cb + kk;
    float w = Wk[((cout * 64 + cin) * 3 + ky) * 3 + kx];
    unsigned short hb = f2bf_rne(w);
    unsigned short lb = f2bf_rne(w - bf2f(hb));
    AHI[obase + j] = __builtin_bit_cast(__bf16, hb);
    ALO[obase + j] = __builtin_bit_cast(__bf16, lb);
  }
}

// ---------------------------------------------------------------------------
// Kernel 1: K = conv3x3(x, Wk) + bk  via implicit-GEMM WMMA.
// One workgroup = one (image n, output row y): 64 cout x 64 pixels.
// 8 waves; wave w -> mtile = w>>1, pixel tiles {2*(w&1), 2*(w&1)+1}.
// x staged in LDS as [3 rows][66 cols][64 cin] bf16 hi/lo.
// ---------------------------------------------------------------------------
__global__ __launch_bounds__(256) void conv_k_wmma(
    const float* __restrict__ x, const __bf16* __restrict__ AHI,
    const __bf16* __restrict__ ALO, const float* __restrict__ bk,
    float* __restrict__ Kbuf) {
  __shared__ __align__(16) __bf16 XHI[3 * 66 * 64];
  __shared__ __align__(16) __bf16 XLO[3 * 66 * 64];

  const int n  = blockIdx.x >> 6;
  const int y0 = blockIdx.x & 63;
  const int tid = threadIdx.x;

  // Stage halo tile: iterate [dy][cin][xx] so global reads coalesce along x.
  for (int e = tid; e < 3 * 64 * 66; e += 256) {
    int dy  = e / (64 * 66);
    int rem = e - dy * (64 * 66);
    int cin = rem / 66;
    int xx  = rem - cin * 66;
    int gy = y0 + dy - 1;
    int gx = xx - 1;
    float v = 0.f;
    if ((unsigned)gy < 64u && (unsigned)gx < 64u)
      v = x[(((n * 64 + cin) << 6) + gy) * 64 + gx];
    unsigned short hb = f2bf_rne(v);
    unsigned short lb = f2bf_rne(v - bf2f(hb));
    int li = (dy * 66 + xx) * 64 + cin;
    XHI[li] = __builtin_bit_cast(__bf16, hb);
    XLO[li] = __builtin_bit_cast(__bf16, lb);
  }
  __syncthreads();

  const int wave  = tid >> 5;
  const int lane  = tid & 31;
  const int lhalf = lane >> 4;   // which 16-lane half
  const int ln    = lane & 15;

  const int mtile  = wave >> 1;            // cout tile (== head)
  const int ntile0 = (wave & 1) * 2;       // first of two pixel tiles
  const int px0    = ntile0 * 16 + ln;
  const int px1    = px0 + 16;

  // C layout: VGPR r -> M = r + 8*lhalf, N = ln.  Seed with bias.
  v8f acc0, acc1;
#pragma unroll
  for (int r = 0; r < 8; ++r) {
    float b = bk[mtile * 16 + r + lhalf * 8];
    acc0[r] = b;
    acc1[r] = b;
  }

#pragma unroll
  for (int tap = 0; tap < 9; ++tap) {
    const int ky = tap / 3, kx = tap % 3;
    const int xl0 = px0 + kx;                     // LDS x index (halo built in)
    const int xl1 = px1 + kx;
#pragma unroll
    for (int half = 0; half < 2; ++half) {
      const int t = tap * 2 + half;
      // A fragments: one 32B coalesced load per lane, shared by BOTH tiles.
      const v8u* pah = (const v8u*)(AHI + ((mtile * KSTEPS + t) * 32 + lane) * 16);
      const v8u* pal = (const v8u*)(ALO + ((mtile * KSTEPS + t) * 32 + lane) * 16);
      v16bf Ah = __builtin_bit_cast(v16bf, pah[0]);
      v16bf Al = __builtin_bit_cast(v16bf, pal[0]);
      // B fragments (32x16 bf16): lanes 0-15 hold K={0..7,16..23}, lanes 16-31
      // K={8..15,24..31} -> per lane two aligned 16B loads.
      const int cb2 = half * 32 + lhalf * 8;
      const int lb0 = (ky * 66 + xl0) * 64 + cb2;
      const int lb1 = (ky * 66 + xl1) * 64 + cb2;
      v16bf Bh0 = __builtin_bit_cast(v16bf, ldB(XHI + lb0));
      v16bf Bl0 = __builtin_bit_cast(v16bf, ldB(XLO + lb0));
      v16bf Bh1 = __builtin_bit_cast(v16bf, ldB(XHI + lb1));
      v16bf Bl1 = __builtin_bit_cast(v16bf, ldB(XLO + lb1));
      // bf16x3 split: hi*hi + hi*lo + lo*hi (~fp32 accuracy, f32 accumulate)
      acc0 = __builtin_amdgcn_wmma_f32_16x16x32_bf16(false, Ah, false, Bh0,
                                                     (short)0, acc0, false, false);
      acc1 = __builtin_amdgcn_wmma_f32_16x16x32_bf16(false, Ah, false, Bh1,
                                                     (short)0, acc1, false, false);
      acc0 = __builtin_amdgcn_wmma_f32_16x16x32_bf16(false, Ah, false, Bl0,
                                                     (short)0, acc0, false, false);
      acc1 = __builtin_amdgcn_wmma_f32_16x16x32_bf16(false, Ah, false, Bl1,
                                                     (short)0, acc1, false, false);
      acc0 = __builtin_amdgcn_wmma_f32_16x16x32_bf16(false, Al, false, Bh0,
                                                     (short)0, acc0, false, false);
      acc1 = __builtin_amdgcn_wmma_f32_16x16x32_bf16(false, Al, false, Bh1,
                                                     (short)0, acc1, false, false);
    }
  }
#pragma unroll
  for (int r = 0; r < 8; ++r) {
    int cout = mtile * 16 + r + lhalf * 8;
    float* row = Kbuf + (((n * 64 + cout) << 6) + y0) * 64;
    row[px0] = acc0[r];
    row[px1] = acc1[r];
  }
}

// ---------------------------------------------------------------------------
// Kernel 2: sk = conv3x3(K[head channels], Ws key-half), no bias (cancels in
// softmax).  Tiny (~0.6 GFLOP).  SK stored [head][b][y*64+x][s] so kernel 3
// reads the 32 softmax inputs contiguously.
// ---------------------------------------------------------------------------
__global__ __launch_bounds__(64) void sk_conv(const float* __restrict__ Kbuf,
                                              const float* __restrict__ Ws,
                                              float* __restrict__ SK) {
  int bid  = blockIdx.x;
  int y    = bid & 63;
  int n    = (bid >> 6) & 127;
  int head = bid >> 13;
  int xcol = threadIdx.x;

  __shared__ float w[16 * 9];
  for (int i = threadIdx.x; i < 144; i += 64)
    w[i] = Ws[(head * 32 + 16 + i / 9) * 9 + (i % 9)];   // key half of Ws
  __syncthreads();

  float acc = 0.f;
#pragma unroll
  for (int c = 0; c < 16; ++c)
#pragma unroll
    for (int ky = 0; ky < 3; ++ky)
#pragma unroll
      for (int kx = 0; kx < 3; ++kx) {
        int gy = y + ky - 1, gx = xcol + kx - 1;
        if ((unsigned)gy < 64u && (unsigned)gx < 64u)
          acc += w[c * 9 + ky * 3 + kx] *
                 Kbuf[(((n * 64 + head * 16 + c) << 6) + gy) * 64 + gx];
      }
  int b = n >> 5, s = n & 31;
  SK[((((head * 4 + b) << 12) + (y << 6) + xcol) << 5) + s] = acc;
}

// ---------------------------------------------------------------------------
// Kernel 3: softmax over keys + weighted sum; broadcast over all 32 query
// positions (exactly equal by the softmax factorization).
// ---------------------------------------------------------------------------
__global__ __launch_bounds__(64) void attn_out(const float* __restrict__ Kbuf,
                                               const float* __restrict__ SK,
                                               float* __restrict__ out) {
  int bid  = blockIdx.x;              // (head, b, y)
  int y    = bid & 63;
  int b    = (bid >> 6) & 3;
  int head = bid >> 8;
  int xcol = threadIdx.x;

  const float* sp = SK + ((((head * 4 + b) << 12) + (y << 6) + xcol) << 5);
  float w[32];
  float m = -3.0e38f;
#pragma unroll
  for (int s = 0; s < 32; ++s) { w[s] = sp[s]; m = fmaxf(m, w[s]); }
  float sum = 0.f;
#pragma unroll
  for (int s = 0; s < 32; ++s) { w[s] = expf(w[s] - m); sum += w[s]; }
  float inv = 1.f / sum;
#pragma unroll
  for (int s = 0; s < 32; ++s) w[s] *= inv;

#pragma unroll
  for (int c = 0; c < 16; ++c) {
    float acc = 0.f;
#pragma unroll
    for (int s = 0; s < 32; ++s)
      acc += w[s] * Kbuf[((((b * 32 + s) * 64 + head * 16 + c) << 6) + y) * 64 + xcol];
#pragma unroll
    for (int j = 0; j < 32; ++j)
      out[((((b * 32 + j) * 64 + head * 16 + c) << 6) + y) * 64 + xcol] = acc;
  }
}

// ---------------------------------------------------------------------------
// Workspace layout (bytes):
//   [0,         73728)   AHI  (4*18*32*16 bf16)
//   [73728,    147456)   ALO
//   [147456, +128 MiB)   Kbuf (128*64*64*64 f32)
//   then        8 MiB    SK   (4*4*4096*32 f32)
// total ~136.1 MiB
// ---------------------------------------------------------------------------
extern "C" void kernel_launch(void* const* d_in, const int* in_sizes, int n_in,
                              void* d_out, int out_size, void* d_ws, size_t ws_size,
                              hipStream_t stream) {
  (void)in_sizes; (void)n_in; (void)out_size; (void)ws_size;
  const float* x  = (const float*)d_in[0];
  const float* Wk = (const float*)d_in[3];
  const float* bk = (const float*)d_in[4];
  const float* Ws = (const float*)d_in[5];
  // d_in[1] (Wq), d_in[2] (bq), d_in[6] (bs) provably cancel in the softmax.

  char*   ws   = (char*)d_ws;
  __bf16* AHI  = (__bf16*)(ws);
  __bf16* ALO  = (__bf16*)(ws + 73728);
  float*  Kbuf = (float*)(ws + 147456);
  float*  SK   = (float*)(ws + 147456 + (size_t)134217728);
  float*  out  = (float*)d_out;

  wk_rearrange<<<(4 * KSTEPS * 32 + 255) / 256, 256, 0, stream>>>(Wk, AHI, ALO);
  conv_k_wmma<<<NIMG * HW, 256, 0, stream>>>(x, AHI, ALO, bk, Kbuf);
  sk_conv<<<NHEADS * NIMG * HW, 64, 0, stream>>>(Kbuf, Ws, SK);
  attn_out<<<NHEADS * NB * HW, 64, 0, stream>>>(Kbuf, SK, out);
}
